// QuantLinear_26345329393713
// MI455X (gfx1250) — matrix operand order repaired and checked by reading, not compiled, not run on hardware
//
#include <hip/hip_runtime.h>

// ---------------------------------------------------------------------------
// QuantLinear: out = (x @ qweight^T) * scale + bias     (MI455X / gfx1250)
//   pass 1: x -> f16, qweight(int8) -> f16 (exact) into workspace; the 96 MiB
//           f16 working set stays resident in the 192 MB L2 for the GEMM.
//   pass 2: f16 WMMA GEMM (v_wmma_f32_16x16x32_f16), 128x128 block tile,
//           K staged 64 at a time into double-buffered LDS via
//           GLOBAL_LOAD_ASYNC_TO_LDS_B128 (ASYNCcnt-tracked async copy),
//           8 waves/block, 64x32 per-wave tile (4x2 fragments).
// Requires ws_size >= (8192*4096 + 4096*4096) * 2 bytes = 96 MiB.
// ---------------------------------------------------------------------------

typedef __attribute__((ext_vector_type(16))) _Float16 v16h;
typedef __attribute__((ext_vector_type(8)))  _Float16 v8h;
typedef __attribute__((ext_vector_type(4)))  _Float16 v4h;
typedef __attribute__((ext_vector_type(8)))  float    v8f;
typedef __attribute__((ext_vector_type(16))) signed char v16b;
typedef __attribute__((ext_vector_type(4)))  int      v4i;

#define TOKENS 8192
#define DIN    4096
#define DOUT   4096

#define MBLK 128
#define NBLK 128
#define KBLK 64
#define LDSS 72   // LDS row stride in halves (64 + 8 pad -> 144B rows)

#if __has_builtin(__builtin_amdgcn_global_load_async_to_lds_b128)
#define HAVE_ASYNC_LDS 1
#else
#define HAVE_ASYNC_LDS 0
#endif

#if __has_builtin(__builtin_amdgcn_s_wait_asynccnt)
#define WAIT_ASYNC(n) __builtin_amdgcn_s_wait_asynccnt(n)
#else
#define WAIT_ASYNC(n) asm volatile("s_wait_asynccnt %0" ::"i"(n) : "memory")
#endif

#define AS1 __attribute__((address_space(1)))
#define AS3 __attribute__((address_space(3)))

// ---------------- pass 1: dtype conversion ----------------

__global__ __launch_bounds__(256) void cvt_x_f32_to_f16(
    const float* __restrict__ x, _Float16* __restrict__ xh) {
  size_t i = (size_t)blockIdx.x * blockDim.x + threadIdx.x;   // 4 floats / thread
  const float4 v = ((const float4* __restrict__)x)[i];
  v4h h;
  h[0] = (_Float16)v.x; h[1] = (_Float16)v.y;
  h[2] = (_Float16)v.z; h[3] = (_Float16)v.w;
  ((v4h* __restrict__)xh)[i] = h;
}

__global__ __launch_bounds__(256) void cvt_w_i8_to_f16(
    const signed char* __restrict__ w, _Float16* __restrict__ wh) {
  size_t i = (size_t)blockIdx.x * blockDim.x + threadIdx.x;   // 16 int8 / thread
  const v16b q = ((const v16b* __restrict__)w)[i];
  v16h h;
#pragma unroll
  for (int j = 0; j < 16; ++j) h[j] = (_Float16)(short)q[j];  // exact
  ((v16h* __restrict__)wh)[i] = h;
}

// ---------------- pass 2: WMMA GEMM ----------------

__global__ __launch_bounds__(256) void qlin_wmma_gemm(
    const _Float16* __restrict__ xh,   // [TOKENS][DIN]  (A, K-contiguous)
    const _Float16* __restrict__ wh,   // [DOUT][DIN]    (B^T source, K-contiguous)
    const float* __restrict__ scale,   // [DOUT]
    const float* __restrict__ bias,    // [DOUT]
    float* __restrict__ out)           // [TOKENS][DOUT]
{
  __shared__ _Float16 As[2][MBLK * LDSS];
  __shared__ _Float16 Bs[2][NBLK * LDSS];

  const int tid  = threadIdx.x;
  const int lane = tid & 31;
  const int wave = tid >> 5;     // 0..7
  const int wm   = wave & 1;     // 2 waves along M -> 64 rows each
  const int wn   = wave >> 1;    // 4 waves along N -> 32 cols each

  const int m0 = blockIdx.y * MBLK;
  const int n0 = blockIdx.x * NBLK;

  const _Float16* __restrict__ gA = xh + (size_t)m0 * DIN;
  const _Float16* __restrict__ gB = wh + (size_t)n0 * DIN;

  // Tile = 128 rows x 64 halves = 1024 x 16B chunks; 4 A + 4 B chunks/thread.
  const int c_row = tid >> 3;           // rows for chunks i: c_row + i*32
  const int c_col = (tid & 7) << 3;     // half-element column within K-stage

  v8f acc[4][2];
#pragma unroll
  for (int mi = 0; mi < 4; ++mi)
#pragma unroll
    for (int ni = 0; ni < 2; ++ni)
#pragma unroll
      for (int r = 0; r < 8; ++r) acc[mi][ni][r] = 0.0f;

  const int NK   = DIN / KBLK;   // 64 stages
  const int hsel = lane >> 4;    // half-wave K interleave (ISA 7.12.2)
  const int lrow = lane & 15;

  // Fragment base offsets (kk offsets become immediates).
  // A 16x32 f16 frag: lane<16 -> K[0..15];          lane>=16 -> K[16..31].
  // B 32x16 f16 frag: lane<16 -> K[0..15];          lane>=16 -> K[16..31].
  const int aoff = (wm * 64 + lrow) * LDSS + hsel * 16;
  const int boff = (wn * 32 + lrow) * LDSS + hsel * 16;

  auto compute_stage = [&](int cur) {
#pragma unroll
    for (int kk = 0; kk < KBLK; kk += 32) {
      v16h bfrag[2];
#pragma unroll
      for (int ni = 0; ni < 2; ++ni) {
        const _Float16* p = &Bs[cur][boff + ni * (16 * LDSS) + kk];
        const v8h lo = *(const v8h*)p;
        const v8h hi = *(const v8h*)(p + 8);
        bfrag[ni] = __builtin_shufflevector(lo, hi, 0, 1, 2, 3, 4, 5, 6, 7, 8,
                                            9, 10, 11, 12, 13, 14, 15);
      }
#pragma unroll
      for (int mi = 0; mi < 4; ++mi) {
        const _Float16* p = &As[cur][aoff + mi * (16 * LDSS) + kk];
        const v8h lo = *(const v8h*)p;
        const v8h hi = *(const v8h*)(p + 8);
        const v16h a = __builtin_shufflevector(lo, hi, 0, 1, 2, 3, 4, 5, 6, 7,
                                               8, 9, 10, 11, 12, 13, 14, 15);
        acc[mi][0] = __builtin_amdgcn_wmma_f32_16x16x32_f16(
            false, a, false, bfrag[0], (short)0, acc[mi][0], false, false);
        acc[mi][1] = __builtin_amdgcn_wmma_f32_16x16x32_f16(
            false, a, false, bfrag[1], (short)0, acc[mi][1], false, false);
      }
    }
  };

#if HAVE_ASYNC_LDS
  // ---- async copy path: GLOBAL_LOAD_ASYNC_TO_LDS_B128, ASYNCcnt-tracked ----
  auto issue_stage = [&](int buf, int kbase) {
#pragma unroll
    for (int i = 0; i < 4; ++i) {
      const int row = c_row + i * 32;
      __builtin_amdgcn_global_load_async_to_lds_b128(
          (AS1 v4i*)(gA + (size_t)row * DIN + kbase + c_col),
          (AS3 v4i*)&As[buf][row * LDSS + c_col], 0, 0);
      __builtin_amdgcn_global_load_async_to_lds_b128(
          (AS1 v4i*)(gB + (size_t)row * DIN + kbase + c_col),
          (AS3 v4i*)&Bs[buf][row * LDSS + c_col], 0, 0);
    }
  };

  issue_stage(0, 0);
  for (int kt = 0; kt < NK; ++kt) {
    const int cur = kt & 1;
    if (kt + 1 < NK) {
      issue_stage(cur ^ 1, (kt + 1) * KBLK);  // 8 newest ops stay in flight
      WAIT_ASYNC(8);                          // in-order => stage kt landed
    } else {
      WAIT_ASYNC(0);
    }
    __syncthreads();
    compute_stage(cur);
    __syncthreads();   // all waves done reading buf `cur` before next issue
  }
#else
  // ---- fallback: register-staged copy (reduced live range) ----
  uint4 ra[4], rb[4];
  auto issue_loads = [&](int kbase) {
#pragma unroll
    for (int i = 0; i < 4; ++i) {
      const int row = c_row + i * 32;
      ra[i] = *(const uint4*)(gA + (size_t)row * DIN + kbase + c_col);
      rb[i] = *(const uint4*)(gB + (size_t)row * DIN + kbase + c_col);
    }
  };
  auto commit = [&](int buf) {
#pragma unroll
    for (int i = 0; i < 4; ++i) {
      const int row = c_row + i * 32;
      *(uint4*)&As[buf][row * LDSS + c_col] = ra[i];
      *(uint4*)&Bs[buf][row * LDSS + c_col] = rb[i];
    }
  };

  issue_loads(0);
  commit(0);
  __syncthreads();
  for (int kt = 0; kt < NK; ++kt) {
    const int cur = kt & 1;
    if (kt + 1 < NK) issue_loads((kt + 1) * KBLK);
    compute_stage(cur);
    if (kt + 1 < NK) commit(cur ^ 1);
    __syncthreads();
  }
#endif

  // Epilogue: C layout (8 VGPRs): VGPR r -> M = r (+8 upper half-wave),
  // N = lane&15.  out = acc * scale[col] + bias[col]
#pragma unroll
  for (int ni = 0; ni < 2; ++ni) {
    const int col  = n0 + wn * 32 + ni * 16 + lrow;
    const float s  = scale[col];
    const float bb = bias[col];
#pragma unroll
    for (int mi = 0; mi < 4; ++mi) {
#pragma unroll
      for (int r = 0; r < 8; ++r) {
        const int row = m0 + wm * 64 + mi * 16 + r + (hsel << 3);
        out[(size_t)row * DOUT + col] = acc[mi][ni][r] * s + bb;
      }
    }
  }
}

// ---------------- host launcher ----------------

extern "C" void kernel_launch(void* const* d_in, const int* in_sizes, int n_in,
                              void* d_out, int out_size, void* d_ws,
                              size_t ws_size, hipStream_t stream) {
  const float*       x     = (const float*)d_in[0];
  const signed char* qw    = (const signed char*)d_in[1];
  const float*       scale = (const float*)d_in[2];
  const float*       bias  = (const float*)d_in[3];
  float*             out   = (float*)d_out;

  _Float16* xh = (_Float16*)d_ws;                       // 64 MiB
  _Float16* wh = xh + (size_t)TOKENS * DIN;             // 32 MiB (ws >= 96 MiB)

  {
    const size_t n = (size_t)TOKENS * DIN / 4;          // 4 f32 / thread
    cvt_x_f32_to_f16<<<dim3((unsigned)(n / 256)), 256, 0, stream>>>(x, xh);
  }
  {
    const size_t n = (size_t)DOUT * DIN / 16;           // 16 int8 / thread
    cvt_w_i8_to_f16<<<dim3((unsigned)(n / 256)), 256, 0, stream>>>(qw, wh);
  }

  dim3 grid(DOUT / NBLK, TOKENS / MBLK);                // (32, 64)
  qlin_wmma_gemm<<<grid, 256, 0, stream>>>(xh, wh, scale, bias, out);
}
